// PostProcessor_65850438582778
// MI455X (gfx1250) — matrix operand order repaired and checked by reading, not compiled, not run on hardware
//
#include <hip/hip_runtime.h>
#include <hip/hip_bf16.h>

typedef __attribute__((ext_vector_type(16))) __bf16 v16bf;
typedef __attribute__((ext_vector_type(8)))  float  v8f;

#define B_   16
#define CIN  128
#define CH   256
#define H_   96
#define W_   96
#define S_   (H_*W_)          // 9216 spatial per batch
#define NT   128              // spatial tile per workgroup
#define XS   (CIN + 8)        // padded LDS stride for x tile (bf16 elems)
#define HS   (CH + 8)         // padded LDS stride for h1/h2 (bf16 elems)
#define REGION0   (NT*HS*2)   // 67584 B : holds XB first, then H2
#define LDS_BYTES (2*NT*HS*2) // 135168 B dynamic LDS

// ---------------------------------------------------------------------------
// Pack a row-major OxC f32 weight matrix into WMMA bf16 A-fragment order:
// fragment f = m*KT + kt is 32 lanes x 16 bf16 contiguous (32 B per lane).
// Lane L holds M = m*16 + (L&15); element e -> K = kt*32 + (L>>4)*16 + e
//   (contiguous k order, matching the B fragment reads)
// ---------------------------------------------------------------------------
__global__ void pack_wmma_a(const float* __restrict__ Wm,
                            __bf16* __restrict__ out, int O, int C) {
  int KT   = C >> 5;
  int frag = blockIdx.x;            // m*KT + kt
  int m    = frag / KT;
  int kt   = frag % KT;
  int lane = threadIdx.x;           // 0..31
  int M    = m*16 + (lane & 15);
  int kb   = kt*32 + (lane >> 4)*16;
  __bf16* dst = out + ((size_t)frag*32 + lane)*16;
  #pragma unroll
  for (int e = 0; e < 8; ++e) {
    dst[e]     = (__bf16)Wm[(size_t)M*C + kb + e];
    dst[e + 8] = (__bf16)Wm[(size_t)M*C + kb + 8 + e];
  }
}

// 32-byte fragment load via two 16-byte chunks (16B-aligned LDS/global safe)
__device__ __forceinline__ v16bf ld_frag16(const __bf16* p) {
  union { uint4 u[2]; v16bf v; } c;
  c.u[0] = *(const uint4*)(p);
  c.u[1] = *(const uint4*)(p + 8);
  return c.v;
}

// 32-byte bias vector load (8 consecutive f32)
__device__ __forceinline__ v8f ld_f8(const float* p) {
  union { uint4 u[2]; v8f v; } c;
  c.u[0] = *(const uint4*)(p);
  c.u[1] = *(const uint4*)(p + 4);
  return c.v;
}

// Pack 8 f32 accumulators (one D-fragment row-block) -> 8 bf16, one 16B store
__device__ __forceinline__ void st_bf8(__bf16* p, v8f f, bool relu) {
  union { __bf16 h[8]; uint4 u; } c;
  #pragma unroll
  for (int r = 0; r < 8; ++r) {
    float v = f[r];
    if (relu) v = fmaxf(v, 0.0f);
    c.h[r] = (__bf16)v;
  }
  *(uint4*)p = c.u;
}

// ---------------------------------------------------------------------------
// One GEMM stage: D(MTILES*16 x 128) = act(Wp @ Bsrc + bias) -> Dst (LDS bf16)
// Each wave owns a fixed M-row: its A fragments are loaded from global ONCE
// and kept in registers; B fragments stream from LDS (2 ds_load_b128 / wmma).
// ---------------------------------------------------------------------------
template <int MTILES, int KTILES, bool RELU>
__device__ __forceinline__ void gemm_stage(const __bf16* __restrict__ wp,
                                           const float*  __restrict__ bias,
                                           const __bf16* __restrict__ Bsrc, int bstride,
                                           __bf16* __restrict__ Dst, int dstride,
                                           int wave, int lane) {
  int row  = lane & 15;
  int half = lane >> 4;
  #pragma unroll
  for (int mi = 0; mi < MTILES/8; ++mi) {
    int mt = mi*8 + wave;
    v16bf a[KTILES];
    #pragma unroll
    for (int k = 0; k < KTILES; ++k)
      a[k] = ld_frag16(wp + ((size_t)(mt*KTILES + k)*32 + lane)*16);
    v8f bv = ld_f8(bias + mt*16 + half*8);
    #pragma unroll
    for (int n = 0; n < 8; ++n) {
      v8f acc = bv;
      #pragma unroll
      for (int k = 0; k < KTILES; ++k) {
        v16bf b = ld_frag16(Bsrc + (n*16 + row)*bstride + k*32 + half*16);
        acc = __builtin_amdgcn_wmma_f32_16x16x32_bf16(
                false, a[k], false, b, (short)0, acc, false, false);
      }
      st_bf8(Dst + (n*16 + row)*dstride + mt*16 + half*8, acc, RELU);
    }
  }
}

// ---------------------------------------------------------------------------
// Fused conv1->relu->conv2->relu->conv3 GEMM chain. One block = 128 spatial
// columns of one batch; all intermediates stay in LDS as bf16.
// ---------------------------------------------------------------------------
__global__ void __launch_bounds__(256)
gemm_chain(const float* __restrict__ x,
           const __bf16* __restrict__ w1p, const float* __restrict__ b1,
           const __bf16* __restrict__ w2p, const float* __restrict__ b2,
           const __bf16* __restrict__ w3p, const float* __restrict__ b3,
           float* __restrict__ ox) {
  extern __shared__ char smem[];
  __bf16* XB = (__bf16*)smem;               // [NT][XS], dead after conv1
  __bf16* H2 = (__bf16*)smem;               // [NT][HS], aliases XB region
  __bf16* H1 = (__bf16*)(smem + REGION0);   // [NT][HS]

  int blk = blockIdx.x;
  int b   = blk / (S_/NT);
  int s0  = (blk % (S_/NT)) * NT;
  int tid = threadIdx.x, wave = tid >> 5, lane = tid & 31;

  // warm L2 for the big conv2 weight array while we stage x
  __builtin_prefetch(w2p + (size_t)tid * 256, 0, 0);

  // stage x tile -> LDS bf16, [n][c] layout (coalesced global reads)
  for (int idx = tid; idx < NT*CIN; idx += 256) {
    int n = idx & (NT - 1);
    int c = idx >> 7;
    XB[n*XS + c] = (__bf16)x[((size_t)(b*CIN + c))*S_ + s0 + n];
  }
  __syncthreads();

  gemm_stage<16, 4, true>(w1p, b1, XB, XS, H1, HS, wave, lane);  // 128->256
  __syncthreads();
  gemm_stage<16, 8, true>(w2p, b2, H1, HS, H2, HS, wave, lane);  // 256->256
  __syncthreads();

  // conv3: 256->128, f32 result straight to global ox. One M-row per wave;
  // A fragments register-resident, B streams from LDS.
  {
    int row = lane & 15, half = lane >> 4;
    int mt = wave;                       // 8 M-tiles, 8 waves
    v16bf a[8];
    #pragma unroll
    for (int k = 0; k < 8; ++k)
      a[k] = ld_frag16(w3p + ((size_t)(mt*8 + k)*32 + lane)*16);
    v8f bv = ld_f8(b3 + mt*16 + half*8);
    #pragma unroll
    for (int n = 0; n < 8; ++n) {
      v8f acc = bv;
      #pragma unroll
      for (int k = 0; k < 8; ++k) {
        v16bf bb = ld_frag16(H2 + (n*16 + row)*HS + k*32 + half*16);
        acc = __builtin_amdgcn_wmma_f32_16x16x32_bf16(
                false, a[k], false, bb, (short)0, acc, false, false);
      }
      int col = s0 + n*16 + row;
      #pragma unroll
      for (int r = 0; r < 8; ++r)
        ox[((size_t)(b*CIN + mt*16 + half*8 + r))*S_ + col] = acc[r];
    }
  }
}

// ---------------------------------------------------------------------------
// offsets = conv3x3(ox, off_w) + off_b   (128 -> 2 channels, SAME)
// One block per (b, y); ox rows come from L2 (ox fits in 192MB L2).
// ---------------------------------------------------------------------------
__global__ void offsets_kernel(const float* __restrict__ ox,
                               const float* __restrict__ off_w,
                               const float* __restrict__ off_b,
                               float* __restrict__ offs) {
  int blk = blockIdx.x;
  int b = blk / H_, y = blk % H_;
  int xcol = threadIdx.x;
  if (xcol >= W_) return;
  float a0 = off_b[0], a1 = off_b[1];
  const float* plane0 = ox + (size_t)b*CIN*S_;
  for (int c = 0; c < CIN; ++c) {
    const float* pl  = plane0 + (size_t)c*S_;
    const float* wk0 = off_w + c*9;
    const float* wk1 = off_w + (CIN + c)*9;
    #pragma unroll
    for (int ky = 0; ky < 3; ++ky) {
      int yy = y + ky - 1;
      if (yy < 0 || yy >= H_) continue;
      #pragma unroll
      for (int kx = 0; kx < 3; ++kx) {
        int xx = xcol + kx - 1;
        float v = (xx >= 0 && xx < W_) ? pl[yy*W_ + xx] : 0.0f;
        a0 = fmaf(v, wk0[ky*3 + kx], a0);
        a1 = fmaf(v, wk1[ky*3 + kx], a1);
      }
    }
  }
  offs[((size_t)(b*2 + 0)*H_ + y)*W_ + xcol] = a0;
  offs[((size_t)(b*2 + 1)*H_ + y)*W_ + xcol] = a1;
}

// ---------------------------------------------------------------------------
// out = ox + gumbel + offset_feat + blurred.  One block per (b,c,row i).
// Reference broadcasting quirk: sample row = j + offy[b,i,j], col = i + offx.
// ---------------------------------------------------------------------------
__global__ void finalize_kernel(const float* __restrict__ ox,
                                const float* __restrict__ noise,
                                const float* __restrict__ offs,
                                float* __restrict__ out) {
  __shared__ float rows[5][W_];
  int blk = blockIdx.x;
  int i = blk % H_;
  int c = (blk / H_) % CIN;
  int b = blk / (H_ * CIN);
  const float* plane = ox + ((size_t)(b*CIN + c))*S_;

  for (int t = threadIdx.x; t < 5*W_; t += blockDim.x) {
    int r = t / W_, col = t % W_;
    int y = i + r - 2;
    rows[r][col] = (y >= 0 && y < H_) ? plane[y*W_ + col] : 0.0f;
  }
  __syncthreads();

  int j = threadIdx.x;
  if (j >= W_) return;

  float oxv = rows[2][j];

  // gumbel-sigmoid, TEMP = 0.5
  float u = noise[((size_t)(b*CIN + c))*S_ + i*W_ + j];
  u = fminf(fmaxf(u, 1e-6f), 1.0f - 1e-6f);
  float logistic = logf(u) - log1pf(-u);
  float z = (oxv + logistic) * 2.0f;
  float gum = 1.0f / (1.0f + expf(-z));

  // separable 5x5 gaussian, sigma = 1
  const float g0 = 0.13533528323661270f;  // exp(-2)
  const float g1 = 0.60653065971263342f;  // exp(-0.5)
  const float gk[5] = {g0, g1, 1.0f, g1, g0};
  const float gs  = 1.0f + 2.0f*g1 + 2.0f*g0;
  const float inv = 1.0f / (gs * gs);
  float blur = 0.0f;
  #pragma unroll
  for (int r = 0; r < 5; ++r) {
    float rowacc = 0.0f;
    #pragma unroll
    for (int d = 0; d < 5; ++d) {
      int col = j + d - 2;
      float v = (col >= 0 && col < W_) ? rows[r][col] : 0.0f;
      rowacc = fmaf(v, gk[d], rowacc);
    }
    blur = fmaf(rowacc, gk[r], blur);
  }
  blur *= inv;

  // bilinear gather with zero padding
  float offx = offs[((size_t)(b*2 + 0)*H_ + i)*W_ + j];
  float offy = offs[((size_t)(b*2 + 1)*H_ + i)*W_ + j];
  float ix = (float)i + offx;   // column coordinate
  float iy = (float)j + offy;   // row coordinate
  float x0 = floorf(ix), y0 = floorf(iy);
  float x1 = x0 + 1.0f, y1 = y0 + 1.0f;
  float wa = (x1 - ix)*(y1 - iy), wb = (x1 - ix)*(iy - y0);
  float wc = (ix - x0)*(y1 - iy), wd = (ix - x0)*(iy - y0);
  float feat = 0.0f;
  #pragma unroll
  for (int s = 0; s < 4; ++s) {
    float xf = (s < 2) ? x0 : x1;
    float yf = (s & 1) ? y1 : y0;
    float w  = (s == 0) ? wa : (s == 1) ? wb : (s == 2) ? wc : wd;
    bool valid = (xf >= 0.0f) && (xf <= (float)(W_-1)) &&
                 (yf >= 0.0f) && (yf <= (float)(H_-1));
    int xc = (int)fminf(fmaxf(xf, 0.0f), (float)(W_-1));
    int yc = (int)fminf(fmaxf(yf, 0.0f), (float)(H_-1));
    float v = plane[yc*W_ + xc];
    feat = fmaf(valid ? v : 0.0f, w, feat);
  }

  out[((size_t)(b*CIN + c)*H_ + i)*W_ + j] = oxv + gum + feat + blur;
}

// ---------------------------------------------------------------------------
extern "C" void kernel_launch(void* const* d_in, const int* in_sizes, int n_in,
                              void* d_out, int out_size, void* d_ws, size_t ws_size,
                              hipStream_t stream) {
  const float* x  = (const float*)d_in[0];
  const float* w1 = (const float*)d_in[1];
  const float* b1 = (const float*)d_in[2];
  const float* w2 = (const float*)d_in[3];
  const float* b2 = (const float*)d_in[4];
  const float* w3 = (const float*)d_in[5];
  const float* b3 = (const float*)d_in[6];
  const float* ow = (const float*)d_in[7];
  const float* ob = (const float*)d_in[8];
  const float* nu = (const float*)d_in[9];

  char* ws = (char*)d_ws;
  float*  ox   = (float*)ws;                                        // 75.5 MB
  float*  offs = (float*)(ws + (size_t)B_*CIN*S_*4);                // 1.18 MB
  __bf16* w1p  = (__bf16*)(ws + (size_t)B_*CIN*S_*4 + (size_t)B_*2*S_*4);
  __bf16* w2p  = w1p + (size_t)16*4*512;                            // packed bf16
  __bf16* w3p  = w2p + (size_t)16*8*512;

  pack_wmma_a<<<16*4, 32, 0, stream>>>(w1, w1p, CH,  CIN);
  pack_wmma_a<<<16*8, 32, 0, stream>>>(w2, w2p, CH,  CH);
  pack_wmma_a<<<8*8,  32, 0, stream>>>(w3, w3p, CIN, CH);

  gemm_chain<<<B_*(S_/NT), 256, LDS_BYTES, stream>>>(x, w1p, b1, w2p, b2,
                                                     w3p, b3, ox);
  offsets_kernel<<<B_*H_, 128, 0, stream>>>(ox, ow, ob, offs);
  finalize_kernel<<<B_*CIN*H_, 128, 0, stream>>>(ox, nu, offs, (float*)d_out);
}
